// PointSetAttentionWithEdgeBias_87935160418337
// MI455X (gfx1250) — compile-verified, matrix operand order checked by reading, not run on hardware
//
#include <hip/hip_runtime.h>
#include <hip/hip_bf16.h>
#include <math.h>

// ---------------------------------------------------------------------------
// Problem constants (from reference): N=16384, E=524288 derived at runtime.
// CS=384, CV=16, CZ=128, H=8, CH=16, P=8, PV=8.
// ---------------------------------------------------------------------------

typedef __attribute__((ext_vector_type(16))) _Float16     v16h;
typedef __attribute__((ext_vector_type(8)))  float        v8f;
typedef __attribute__((ext_vector_type(4)))  unsigned int v4u;

union Frag16 { v16h h; v4u u4[2]; };

__device__ __forceinline__ v8f wmma_f32_16x16x32_f16(v16h a, v16h b, v8f c) {
  // 8 args: (neg_a, A, neg_b, B, c_mod, C, reuse_a, reuse_b)
  return __builtin_amdgcn_wmma_f32_16x16x32_f16(false, a, false, b, (short)0, c,
                                                false, false);
}

// order-preserving float <-> uint for atomicMax-based segment max
__device__ __forceinline__ unsigned int orderedF(float f) {
  unsigned int u = __float_as_uint(f);
  return (u & 0x80000000u) ? ~u : (u | 0x80000000u);
}
__device__ __forceinline__ float unorderedF(unsigned int o) {
  return (o & 0x80000000u) ? __uint_as_float(o & 0x7fffffffu)
                           : __uint_as_float(~o);
}

// ---------------------------------------------------------------------------
// Utility kernels
// ---------------------------------------------------------------------------
__global__ void cvt_f32_f16(const float* __restrict__ in, _Float16* __restrict__ out,
                            size_t n) {
  size_t i = (size_t)blockIdx.x * blockDim.x + threadIdx.x;
  if (i < n) out[i] = (_Float16)in[i];
}

__global__ void zero_f32(float* __restrict__ p, size_t n) {
  size_t i = (size_t)blockIdx.x * blockDim.x + threadIdx.x;
  if (i < n) p[i] = 0.0f;
}

// ---------------------------------------------------------------------------
// Fused q/kv projection:  qkv[N,384] = xh[N,384] @ wrows[384,384]^T + bias
// One wave per 16x64 tile; K=384 (12 WMMA steps). Software-pipelined AND
// fully unrolled: buffer rotation becomes register renaming (no v_mov),
// loads for step i+1 overlap the WMMAs of step i (partial s_wait_loadcnt).
// ---------------------------------------------------------------------------
__global__ __launch_bounds__(32) void gemm_qkv_k(
    const _Float16* __restrict__ X, const _Float16* __restrict__ W,
    const float* __restrict__ bq, const float* __restrict__ bkv,
    float* __restrict__ out) {
  const int lane = threadIdx.x & 31;
  const int m    = lane & 15;
  const int half = lane >> 4;
  const int m0   = blockIdx.x << 4;
  const int n0   = blockIdx.y << 6;

  v8f acc[4] = {};
  const _Float16* arow = X + (size_t)(m0 + m) * 384;
  const _Float16* brow[4];
#pragma unroll
  for (int t = 0; t < 4; ++t) brow[t] = W + (size_t)(n0 + t * 16 + m) * 384;

  auto loadf = [&](Frag16& A, Frag16* B, int kb) {
    const int o1 = kb + half * 8;
    const int o2 = kb + 16 + half * 8;
    A.u4[0] = *(const v4u*)(arow + o1);
    A.u4[1] = *(const v4u*)(arow + o2);
#pragma unroll
    for (int t = 0; t < 4; ++t) {
      B[t].u4[0] = *(const v4u*)(brow[t] + o1);
      B[t].u4[1] = *(const v4u*)(brow[t] + o2);
    }
  };

  Frag16 Ac, Bc[4];
  loadf(Ac, Bc, 0);
#pragma unroll
  for (int kb = 32; kb < 384; kb += 32) {
    Frag16 An, Bn[4];
    loadf(An, Bn, kb);  // in flight during the WMMAs below
#pragma unroll
    for (int t = 0; t < 4; ++t) acc[t] = wmma_f32_16x16x32_f16(Ac.h, Bc[t].h, acc[t]);
    Ac = An;
#pragma unroll
    for (int t = 0; t < 4; ++t) Bc[t] = Bn[t];
  }
#pragma unroll
  for (int t = 0; t < 4; ++t) acc[t] = wmma_f32_16x16x32_f16(Ac.h, Bc[t].h, acc[t]);

#pragma unroll
  for (int t = 0; t < 4; ++t) {
    const int col = n0 + t * 16 + m;
    const float bias = (col < 128) ? bq[col] : bkv[col - 128];
#pragma unroll
    for (int j = 0; j < 8; ++j)
      out[(size_t)(m0 + half * 8 + j) * 384 + col] = acc[t][j] + bias;
  }
}

// ---------------------------------------------------------------------------
// Output scalar projection: out_s[N,384] = o[N,128] @ W_out_s[384,128]^T + b
// A converted f32->f16 in-register; K=128 (4 WMMA steps), pipelined+unrolled.
// ---------------------------------------------------------------------------
__global__ __launch_bounds__(32) void gemm_outs_k(
    const float* __restrict__ O, const _Float16* __restrict__ W,
    const float* __restrict__ bias, float* __restrict__ out) {
  const int lane = threadIdx.x & 31;
  const int m    = lane & 15;
  const int half = lane >> 4;
  const int m0   = blockIdx.x << 4;
  const int n0   = blockIdx.y << 6;

  v8f acc[4] = {};
  const float* arow = O + (size_t)(m0 + m) * 128;
  const _Float16* brow[4];
#pragma unroll
  for (int t = 0; t < 4; ++t) brow[t] = W + (size_t)(n0 + t * 16 + m) * 128;

  auto loadf = [&](v16h& ah, Frag16* B, int kb) {
    const int o1 = kb + half * 8;
    const int o2 = kb + 16 + half * 8;
#pragma unroll
    for (int i = 0; i < 8; ++i) {
      ah[i]     = (_Float16)arow[o1 + i];
      ah[8 + i] = (_Float16)arow[o2 + i];
    }
#pragma unroll
    for (int t = 0; t < 4; ++t) {
      B[t].u4[0] = *(const v4u*)(brow[t] + o1);
      B[t].u4[1] = *(const v4u*)(brow[t] + o2);
    }
  };

  v16h Ac = {};
  Frag16 Bc[4];
  loadf(Ac, Bc, 0);
#pragma unroll
  for (int kb = 32; kb < 128; kb += 32) {
    v16h An = {};
    Frag16 Bn[4];
    loadf(An, Bn, kb);
#pragma unroll
    for (int t = 0; t < 4; ++t) acc[t] = wmma_f32_16x16x32_f16(Ac, Bc[t].h, acc[t]);
    Ac = An;
#pragma unroll
    for (int t = 0; t < 4; ++t) Bc[t] = Bn[t];
  }
#pragma unroll
  for (int t = 0; t < 4; ++t) acc[t] = wmma_f32_16x16x32_f16(Ac, Bc[t].h, acc[t]);

#pragma unroll
  for (int t = 0; t < 4; ++t) {
    const int col = n0 + t * 16 + m;
    const float bv = bias[col];
#pragma unroll
    for (int j = 0; j < 8; ++j)
      out[(size_t)(m0 + half * 8 + j) * 384 + col] = acc[t][j] + bv;
  }
}

// ---------------------------------------------------------------------------
// Point projections: tq[N,64,3] (+trans/10) and tkv[N,128,3]
// ---------------------------------------------------------------------------
__global__ void t_proj_k(const float* __restrict__ nv, const float* __restrict__ trans,
                         const float* __restrict__ Wtq, const float* __restrict__ Wtkv,
                         float* __restrict__ tq, float* __restrict__ tkv, int N_) {
  size_t t = (size_t)blockIdx.x * blockDim.x + threadIdx.x;
  if (t >= (size_t)N_ * 192 * 3) return;
  const int n = (int)(t / 576);
  const int rem = (int)(t % 576);
  const int r = rem / 3;
  const int d = rem % 3;
  const float* v = nv + (size_t)n * 48;  // [16 channels][3]
  float s = 0.0f;
  if (r < 64) {
    const float* w = Wtq + r * 16;
#pragma unroll
    for (int c = 0; c < 16; ++c) s += v[c * 3 + d] * w[c];
    s += trans[(size_t)n * 3 + d] * 0.1f;  // trans / D_POINTS
    tq[((size_t)n * 64 + r) * 3 + d] = s;
  } else {
    const float* w = Wtkv + (r - 64) * 16;
#pragma unroll
    for (int c = 0; c < 16; ++c) s += v[c * 3 + d] * w[c];
    tkv[((size_t)n * 128 + (r - 64)) * 3 + d] = s;
  }
}

// ---------------------------------------------------------------------------
// Edge logits + segment max over src. Carries the 268 MB edge-feature stream:
// 32 edges staged per 256-thread block into LDS (conflict-free padded rows),
// killing the 8x per-head read amplification. W_b + softplus(head_weights)
// are also staged once per block. Next tile is prefetched.
// qkv layout: [n][0..127] = q(h*16+c); [n][128 + h*32 + c] = k; +16 = v
// ---------------------------------------------------------------------------
#define EPB 32
__global__ __launch_bounds__(256) void edge_logits_k(
    const float* __restrict__ ef, const int* __restrict__ eidx,
    const float* __restrict__ Wb, const float* __restrict__ bb,
    const float* __restrict__ hwraw, const float* __restrict__ qkv,
    const float* __restrict__ tq, const float* __restrict__ tkv,
    float* __restrict__ logits, unsigned int* __restrict__ amax, int E_) {
  __shared__ __align__(16) float sEf[EPB][132];  // 132: bank-conflict padding
  __shared__ float sWb[8][132];
  __shared__ float sHw[8];
  const int tid = threadIdx.x;
  const size_t ebase = (size_t)blockIdx.x * EPB;

  // stage W_b (8x128) and softplus(head_weights)*sqrt(1/108)
  for (int i = tid; i < 8 * 128; i += 256) sWb[i >> 7][i & 127] = Wb[i];
  if (tid < 8)
    sHw[tid] = logf(1.0f + __expf(hwraw[tid])) * 0.09622504486493764f;

  // stage 32 edges x 128 features with coalesced b128 loads
  {
    const int r  = tid >> 3;       // edge row 0..31
    const int c0 = (tid & 7) * 4;  // 4 float4 chunks per thread
    if (ebase + r < (size_t)E_) {
      const float4* srow = (const float4*)(ef + (ebase + (size_t)r) * 128);
      float4* drow = (float4*)(&sEf[r][0]);
#pragma unroll
      for (int j = 0; j < 4; ++j) drow[c0 + j] = srow[c0 + j];
    }
  }
  // prefetch next tile (global_prefetch_b8) while this one is consumed
  if (ebase + 2 * EPB <= (size_t)E_)
    __builtin_prefetch(ef + (ebase + EPB) * 128 + (size_t)tid * 16, 0, 1);
  __syncthreads();

  const int le = tid >> 3;
  const int h  = tid & 7;
  const size_t e = ebase + le;
  if (e >= (size_t)E_) return;
  const int dst = eidx[e];
  const int src = eidx[E_ + e];

  // edge bias b = ef[e,:] . W_b[h,:] + b_b[h]   (all from LDS)
  float b = bb[h];
#pragma unroll 16
  for (int c = 0; c < 128; ++c) b += sEf[le][c] * sWb[h][c];

  // q[src,h] . k[dst,h]
  const float* qv = qkv + (size_t)src * 384 + h * 16;
  const float* kv = qkv + (size_t)dst * 384 + 128 + h * 32;
  float qk = 0.0f;
#pragma unroll
  for (int c = 0; c < 16; ++c) qk += qv[c] * kv[c];

  // point distance term
  const float* tqs = tq + ((size_t)src * 64 + h * 8) * 3;
  const float* tkd = tkv + ((size_t)dst * 128 + h * 16) * 3;
  float d2 = 0.0f;
#pragma unroll
  for (int i = 0; i < 24; ++i) {
    const float d = tqs[i] - tkd[i];
    d2 += d * d;
  }

  const float a = qk * 0.14433756729740643f                       // sqrt(1/48)
                + 0.57735026918962580f * b                        // sqrt(1/3)
                - 0.5f * d2 * sHw[h];
  logits[e * 8 + h] = a;
  atomicMax(&amax[(size_t)src * 8 + h], orderedF(a));
}

__global__ void edge_exp_k(const int* __restrict__ eidx, const unsigned int* __restrict__ amax,
                           float* __restrict__ logits, float* __restrict__ denom, int E_) {
  size_t t = (size_t)blockIdx.x * blockDim.x + threadIdx.x;
  if (t >= (size_t)E_ * 8) return;
  const int e = (int)(t >> 3);
  const int h = (int)(t & 7);
  const int src = eidx[E_ + e];
  const float mx = unorderedF(amax[(size_t)src * 8 + h]);
  const float ex = __expf(logits[t] - mx);
  logits[t] = ex;
  atomicAdd(&denom[(size_t)src * 8 + h], ex);
}

__global__ void edge_agg_k(const int* __restrict__ eidx, const float* __restrict__ logits,
                           const float* __restrict__ denom, const float* __restrict__ qkv,
                           const float* __restrict__ tkv, float* __restrict__ o,
                           float* __restrict__ opt, int E_) {
  size_t t = (size_t)blockIdx.x * blockDim.x + threadIdx.x;
  if (t >= (size_t)E_ * 8) return;
  const int e = (int)(t >> 3);
  const int h = (int)(t & 7);
  const int dst = eidx[e];
  const int src = eidx[E_ + e];
  const float attn = logits[t] / (denom[(size_t)src * 8 + h] + 1e-16f);

  const float* vv = qkv + (size_t)src * 384 + 128 + h * 32 + 16;  // v[src,h,:]
  float* od = o + (size_t)dst * 128 + h * 16;
#pragma unroll
  for (int c = 0; c < 16; ++c) atomicAdd(&od[c], attn * vv[c]);

  const float* tvs = tkv + ((size_t)src * 128 + h * 16 + 8) * 3;  // t_v[src,h,:,:]
  float* opd = opt + ((size_t)dst * 64 + h * 8) * 3;
#pragma unroll
  for (int i = 0; i < 24; ++i) atomicAdd(&opd[i], attn * tvs[i]);
}

// ---------------------------------------------------------------------------
// out_v[n, cv, d] = sum_j opt[n, j, d] * W_out_v[cv, j]   (j over H*PV=64)
// ---------------------------------------------------------------------------
__global__ void out_v_k(const float* __restrict__ opt, const float* __restrict__ Wov,
                        float* __restrict__ outv, int N_) {
  size_t t = (size_t)blockIdx.x * blockDim.x + threadIdx.x;
  if (t >= (size_t)N_ * 48) return;
  const int n = (int)(t / 48);
  const int rem = (int)(t % 48);
  const int cv = rem / 3;
  const int d = rem % 3;
  const float* op = opt + (size_t)n * 192;
  const float* w = Wov + cv * 64;
  float s = 0.0f;
#pragma unroll 8
  for (int j = 0; j < 64; ++j) s += op[j * 3 + d] * w[j];
  outv[t] = s;
}

// ---------------------------------------------------------------------------
// Host-side orchestration
// ---------------------------------------------------------------------------
extern "C" void kernel_launch(void* const* d_in, const int* in_sizes, int n_in,
                              void* d_out, int out_size, void* d_ws, size_t ws_size,
                              hipStream_t stream) {
  (void)n_in; (void)out_size; (void)ws_size;
  const float* node_scalars  = (const float*)d_in[0];
  const float* node_vectors  = (const float*)d_in[1];
  const float* trans         = (const float*)d_in[2];
  const float* edge_features = (const float*)d_in[3];
  const int*   edge_index    = (const int*)  d_in[4];
  const float* W_q     = (const float*)d_in[5];
  const float* b_q     = (const float*)d_in[6];
  const float* W_kv    = (const float*)d_in[7];
  const float* b_kv    = (const float*)d_in[8];
  const float* W_tq    = (const float*)d_in[9];
  const float* W_tkv   = (const float*)d_in[10];
  const float* W_b     = (const float*)d_in[11];
  const float* b_b     = (const float*)d_in[12];
  const float* head_w  = (const float*)d_in[13];
  const float* W_out_s = (const float*)d_in[14];
  const float* b_out_s = (const float*)d_in[15];
  const float* W_out_v = (const float*)d_in[16];

  const int N = in_sizes[0] / 384;
  const int E = in_sizes[3] / 128;

  char* ws = (char*)d_ws;
  size_t off = 0;
  auto take = [&](size_t bytes) -> char* {
    char* p = ws + off;
    off = (off + bytes + 255) & ~(size_t)255;
    return p;
  };
  _Float16* xh     = (_Float16*)take((size_t)N * 384 * 2);
  _Float16* wrows  = (_Float16*)take((size_t)384 * 384 * 2);
  _Float16* woutsh = (_Float16*)take((size_t)384 * 128 * 2);
  float* qkv    = (float*)take((size_t)N * 384 * 4);
  float* tq     = (float*)take((size_t)N * 64 * 3 * 4);
  float* tkv    = (float*)take((size_t)N * 128 * 3 * 4);
  float* logits = (float*)take((size_t)E * 8 * 4);
  char* zbase   = take((size_t)N * 336 * 4);  // amax | denom | o | opt (zeroed)
  unsigned int* amax = (unsigned int*)zbase;
  float* denom = (float*)(zbase + (size_t)N * 8 * 4);
  float* o     = (float*)(zbase + (size_t)N * 16 * 4);
  float* opt   = (float*)(zbase + (size_t)N * 144 * 4);

  const int thr = 256;
  auto nb = [&](size_t n) -> unsigned int { return (unsigned int)((n + thr - 1) / thr); };

  // 1) precision conversion + accumulator init
  cvt_f32_f16<<<nb((size_t)N * 384), thr, 0, stream>>>(node_scalars, xh, (size_t)N * 384);
  cvt_f32_f16<<<nb((size_t)128 * 384), thr, 0, stream>>>(W_q, wrows, (size_t)128 * 384);
  cvt_f32_f16<<<nb((size_t)256 * 384), thr, 0, stream>>>(W_kv, wrows + (size_t)128 * 384,
                                                         (size_t)256 * 384);
  cvt_f32_f16<<<nb((size_t)384 * 128), thr, 0, stream>>>(W_out_s, woutsh, (size_t)384 * 128);
  zero_f32<<<nb((size_t)N * 336), thr, 0, stream>>>((float*)zbase, (size_t)N * 336);

  // 2) fused q/kv projection (WMMA, software-pipelined + fully unrolled)
  dim3 g1(N / 16, 384 / 64);
  gemm_qkv_k<<<g1, 32, 0, stream>>>(xh, wrows, b_q, b_kv, qkv);

  // 3) point projections
  t_proj_k<<<nb((size_t)N * 576), thr, 0, stream>>>(node_vectors, trans, W_tq, W_tkv,
                                                    tq, tkv, N);

  // 4-6) edge pipeline: logits+max (LDS-tiled), exp+denom, normalized scatter
  edge_logits_k<<<(unsigned int)((E + EPB - 1) / EPB), 256, 0, stream>>>(
      edge_features, edge_index, W_b, b_b, head_w, qkv, tq, tkv, logits, amax, E);
  edge_exp_k<<<nb((size_t)E * 8), thr, 0, stream>>>(edge_index, amax, logits, denom, E);
  edge_agg_k<<<nb((size_t)E * 8), thr, 0, stream>>>(edge_index, logits, denom, qkv, tkv,
                                                    o, opt, E);

  // 7-8) output projections
  float* out_s = (float*)d_out;
  float* out_v = out_s + (size_t)N * 384;
  gemm_outs_k<<<g1, 32, 0, stream>>>(o, woutsh, b_out_s, out_s);
  out_v_k<<<nb((size_t)N * 48), thr, 0, stream>>>(opt, W_out_v, out_v, N);
}
